// CrissCrossAttention_16329465659686
// MI455X (gfx1250) — matrix operand order repaired and checked by reading, not compiled, not run on hardware
//
#include <hip/hip_runtime.h>

// ---------------------------------------------------------------------------
// CDNA5 (gfx1250) criss-cross attention, bf16 WMMA (f32 accumulate).
// B=8, C=256, Cq=32, H=W=128.
// Layouts: x (input)  : [b][c][pix]   (reference layout, fp32 -> bf16 copy)
//          q,k        : [b][pix][32]  (pixel-major, bf16)
//          v          : [b][pix][256] (pixel-major, bf16)
// CDNA5 paths: v_wmma_f32_16x16x32_bf16, tensor_load_to_lds (TDM),
//              global_load_async_to_lds_b128 (ASYNCcnt), s_wait_tensorcnt.
// ---------------------------------------------------------------------------

typedef __attribute__((ext_vector_type(16))) __bf16 v16bf;
typedef __attribute__((ext_vector_type(8)))  __bf16 v8bf;
typedef __attribute__((ext_vector_type(4)))  __bf16 v4bf;
typedef __attribute__((ext_vector_type(8)))  float  v8f;
typedef __attribute__((ext_vector_type(4)))  unsigned v4u;
typedef __attribute__((ext_vector_type(8)))  unsigned v8u;

#define CC_B   8
#define CC_C   256
#define CC_CQ  32
#define CC_H   128
#define CC_W   128
#define CC_HW  (CC_H * CC_W)

// ---- helpers ---------------------------------------------------------------

// Native bf16 convert (v_cvt_pk_bf16_f32, RTE).
static __device__ inline __bf16 f2bf(float f) { return (__bf16)f; }

static __device__ inline unsigned pack2(__bf16 lo, __bf16 hi) {
  unsigned l = (unsigned)__builtin_bit_cast(unsigned short, lo);
  unsigned h = (unsigned)__builtin_bit_cast(unsigned short, hi);
  return l | (h << 16);
}

// D = A*B + C, 16x16x32 bf16, f32 accumulate (codegen-confirmed builtin).
static __device__ inline v8f wmma_bf16(v16bf a, v16bf b, v8f c) {
  return __builtin_amdgcn_wmma_f32_16x16x32_bf16(
      /*neg_a=*/false, a, /*neg_b=*/false, b,
      /*c_mod=*/(short)0, c, /*reuse_a=*/false, /*reuse_b=*/false);
}

// A operand (16-bit A 16x32): lane holds row M=lane&15 with one CONTIGUOUS
// K run: K=0..15 (lanes 0-15) or K=16..31 (lanes 16-31), matching B.
static __device__ inline v16bf load_fragA(const __bf16* base, int koff) {
  v8bf lo = *(const v8bf*)(base + 2 * koff);
  v8bf hi = *(const v8bf*)(base + 2 * koff + 8);
  v16bf f;
#pragma unroll
  for (int e = 0; e < 8; ++e) { f[e] = lo[e]; f[8 + e] = hi[e]; }
  return f;
}

// B operand (order chosen here, same as A): lane holds column N=lane&15 with one CONTIGUOUS
// K run: K=0..15 (lanes 0-15) or K=16..31 (lanes 16-31).
static __device__ inline v16bf load_fragB(const __bf16* base, int boff) {
  return *(const v16bf*)(base + boff);
}

// Async global->LDS copy (16B per lane), tracked by ASYNCcnt (ISA 10.7/15.18).
static __device__ inline void async_ld_b128(void* lds, const void* g) {
  auto l3 = (__attribute__((address_space(3))) char*)lds;
  asm volatile("global_load_async_to_lds_b128 %0, %1, off"
               :: "v"(l3), "v"(g) : "memory");
}
static __device__ inline void async_wait0() {
  asm volatile("s_wait_asynccnt 0x0" ::: "memory");
}

// Tensor Data Mover: 2-D tile load global->LDS (ISA ch.8 D# descriptor).
// tile_d0 contiguous elements per line, tile_d1 lines, line stride in
// data_size (=2B) units.  Issued once per calling wave; EXEC is ignored.
static __device__ inline void tdm_load_2d(void* lds, const void* gsrc,
                                          unsigned tile_d0, unsigned tile_d1,
                                          unsigned stride_elems) {
  unsigned lds_off =
      (unsigned)(uintptr_t)(__attribute__((address_space(3))) char*)lds;
  unsigned long long ga = (unsigned long long)(uintptr_t)gsrc;
  // group0: count=1 | lds_addr | global_addr[56:0] | type=2 (bits 127:126)
  v4u g0 = {1u, lds_off, (unsigned)ga,
            ((unsigned)(ga >> 32) & 0x01FFFFFFu) | 0x80000000u};
  // group1: data_size=1 (2B); tensor_dim0/1 = tile dims (no OOB clipping);
  //         tile_dim0/1; tensor_dim0_stride = stride_elems.
  v8u g1 = {0x00010000u,            // [17:16]=data_size=1 (2 bytes)
            tile_d0 << 16,          // [79:48] tensor_dim0 (low 16 used)
            tile_d1 << 16,          // [111:80] tensor_dim1 (low 16 used)
            tile_d0 << 16,          // [127:112] tile_dim0
            tile_d1,                // [143:128] tile_dim1 ([159:144] td2=0)
            stride_elems,           // [207:160] tensor_dim0_stride (low 32)
            0u, 0u};
  v4u gz = {0u, 0u, 0u, 0u};
  asm volatile("tensor_load_to_lds %0, %1, %2, %3"
               :: "s"(g0), "s"(g1), "s"(gz), "s"(gz) : "memory");
}

// ---- kernel 0: fp32 -> bf16 convert (grid-stride, float4 -> b64) -----------

__global__ void cc_cvt_f32_bf16(const float* __restrict__ in,
                                __bf16* __restrict__ out, int n4) {
  int i = blockIdx.x * blockDim.x + threadIdx.x;
  int stride = gridDim.x * blockDim.x;
  for (; i < n4; i += stride) {
    float4 v = ((const float4*)in)[i];
    __builtin_prefetch(in + (size_t)(i + stride) * 4, 0, 1);
    v4bf o = {f2bf(v.x), f2bf(v.y), f2bf(v.z), f2bf(v.w)};
    *(v4bf*)(out + (size_t)i * 4) = o;
  }
}

// ---- kernel 1: projection GEMM  out[n,m] = sum_k W[m,k]*X[k,n] + bias[m] ---
// grid.x = B * (HW/128)  (one 128-pixel N tile per block, 8 waves x 16 cols)
// grid.y = M chunks of 16*NT rows.  Output is pixel-major [b][n][m].

template <int NT, int MTOT>
__global__ void __launch_bounds__(256)
cc_proj_gemm(const __bf16* __restrict__ Wb, const float* __restrict__ bias,
             const __bf16* __restrict__ xb, __bf16* __restrict__ out) {
  __shared__ __bf16 xsT[128 * 32];                 // X tile, [n][k], 8 KB
  __shared__ __bf16 wls[16 * NT * 256];            // W chunk, [m][k], <=32 KB

  const int b     = (int)blockIdx.x >> 7;
  const int nbase = ((int)blockIdx.x & 127) * 128;
  const int mbase = (int)blockIdx.y * (16 * NT);

  const int tid  = threadIdx.x;
  const int lane = tid & 31;
  const int wave = tid >> 5;
  const int lrow = lane & 15;
  const int koff = (lane < 16) ? 0 : 8;
  const int boff = (lane < 16) ? 0 : 16;
  const int Mhi  = (lane < 16) ? 0 : 8;

  const __bf16* xbat = xb + (size_t)b * CC_C * CC_HW;

  // async-stage the whole W chunk into LDS once (straight copy, 16B/lane)
#pragma unroll
  for (int c = 0; c < 2 * NT; ++c) {
    int off = (tid + c * 256) * 8;                 // element offset
    async_ld_b128(wls + off, Wb + (size_t)mbase * CC_C + off);
  }
  async_wait0();

  v8f acc[NT];
#pragma unroll
  for (int t = 0; t < NT; ++t) acc[t] = {};

#pragma unroll 1
  for (int ks = 0; ks < 8; ++ks) {                 // K = 256 in 8 x 32 steps
    const int kbase = ks * 32;
    // stage transposed X tile: 2 k-rows x 8 n per thread, packed b32 stores
    {
      int kk0 = (tid >> 4) * 2;                    // even k row
      int n0  = (tid & 15) * 8;
      const __bf16* g = xbat + (size_t)(kbase + kk0) * CC_HW + nbase + n0;
      v8bf a8 = *(const v8bf*)g;
      v8bf b8 = *(const v8bf*)(g + CC_HW);
#pragma unroll
      for (int e = 0; e < 8; ++e)
        *(unsigned*)(xsT + (n0 + e) * 32 + kk0) = pack2(a8[e], b8[e]);
    }
    __syncthreads();
    v16bf bfr = load_fragB(&xsT[(wave * 16 + lrow) * 32], boff);
#pragma unroll
    for (int t = 0; t < NT; ++t) {
      v16bf afr = load_fragA(&wls[(t * 16 + lrow) * CC_C + kbase], koff);
      acc[t] = wmma_bf16(afr, bfr, acc[t]);
    }
    __syncthreads();
  }

  // epilogue: pixel-major out -> 8 consecutive m per lane = one b128 store
#pragma unroll
  for (int t = 0; t < NT; ++t) {
    int n = nbase + wave * 16 + lrow;
    int m0 = mbase + t * 16 + Mhi;
    v8bf o8;
#pragma unroll
    for (int r = 0; r < 8; ++r) o8[r] = f2bf(acc[t][r] + bias[m0 + r]);
    *(v8bf*)(out + (size_t)(b * CC_HW + n) * MTOT + m0) = o8;
  }
}

// ---- kernel 2: axial attention along one axis ------------------------------
// MODE 0: column attention (attend over h, fixed w); adds identity to logits;
//         writes out_h (fp32) into outp.
// MODE 1: row attention (attend over w, fixed h); fuses
//         outp = gamma*(outp + out_w) + x.
// One block per (b, fixed) slice; 8 waves, each owns a 16-row i band.

template <int MODE>
__global__ void __launch_bounds__(256)
cc_attn(const __bf16* __restrict__ qb, const __bf16* __restrict__ kb,
        const __bf16* __restrict__ vb, const float* __restrict__ x,
        const float* __restrict__ gptr, float* __restrict__ outp) {
  __shared__ __bf16 att[128 * 128];                // 32 KB, att[i][j]
  __shared__ __bf16 stage[128 * 64];               // 16 KB: qT+kT, then V chunk

  __bf16* qsT = stage;                             // [pos][cq], 4096 elems
  __bf16* ksT = stage + 128 * 32;

  const int b     = (int)blockIdx.x >> 7;
  const int fixed = (int)blockIdx.x & 127;
  // pixel index of position p along the attended axis:
  //   MODE 0: pix = p*W + fixed      MODE 1: pix = fixed*W + p
  const int poff = (MODE == 0) ? fixed : fixed * CC_W;
  const int pstr = (MODE == 0) ? CC_W : 1;

  const int tid  = threadIdx.x;
  const int lane = tid & 31;
  const int wave = tid >> 5;
  const int lrow = lane & 15;
  const int koff = (lane < 16) ? 0 : 8;
  const int boff = (lane < 16) ? 0 : 16;
  const int Mhi  = (lane < 16) ? 0 : 8;
  const int myi  = wave * 16;

  // stage Q^T / K^T slices [pos][cq] via Tensor Data Mover (one wave issues;
  // 128 lines x 32 contiguous bf16, line stride pstr*32 elements)
  {
    const size_t qoff = ((size_t)b * CC_HW + poff) * CC_CQ;
    if (tid == 0) {
      tdm_load_2d(qsT, qb + qoff, CC_CQ, 128, (unsigned)(pstr * CC_CQ));
      tdm_load_2d(ksT, kb + qoff, CC_CQ, 128, (unsigned)(pstr * CC_CQ));
      __builtin_amdgcn_s_wait_tensorcnt(0);
    }
  }
  __syncthreads();

  // logits: S[i,j] = sum_cq Q[cq,i] K[cq,j]   (K-dim = 32, one WMMA per tile)
  v16bf afr = load_fragA(&qsT[(myi + lrow) * 32], koff);
  v8f p[8];
#pragma unroll
  for (int jt = 0; jt < 8; ++jt) {
    v16bf bfr = load_fragB(&ksT[(jt * 16 + lrow) * 32], boff);
    v8f z = {};
    p[jt] = wmma_bf16(afr, bfr, z);
  }

  if (MODE == 0) {                                 // + identity (per source)
#pragma unroll
    for (int jt = 0; jt < 8; ++jt)
#pragma unroll
      for (int r = 0; r < 8; ++r)
        if (myi + r + Mhi == jt * 16 + lrow) p[jt][r] += 1.0f;
  }

  // softmax over j (rows live in 16-lane halves; xor masks 1..8 stay inside)
#pragma unroll
  for (int r = 0; r < 8; ++r) {
    float m = -3.0e38f;
#pragma unroll
    for (int jt = 0; jt < 8; ++jt) m = fmaxf(m, p[jt][r]);
#pragma unroll
    for (int off = 8; off >= 1; off >>= 1) m = fmaxf(m, __shfl_xor(m, off, 32));
    float s = 0.f;
#pragma unroll
    for (int jt = 0; jt < 8; ++jt) {
      float e = __expf(p[jt][r] - m);
      p[jt][r] = e; s += e;
    }
#pragma unroll
    for (int off = 8; off >= 1; off >>= 1) s += __shfl_xor(s, off, 32);
    float inv = 1.0f / s;
#pragma unroll
    for (int jt = 0; jt < 8; ++jt) p[jt][r] *= inv;
  }

  // store attention (bf16) as A operand: att[i][j], j contiguous
#pragma unroll
  for (int jt = 0; jt < 8; ++jt)
#pragma unroll
    for (int r = 0; r < 8; ++r)
      att[(myi + r + Mhi) * 128 + jt * 16 + lrow] = f2bf(p[jt][r]);
  __syncthreads();                                 // att done; stage reusable

  const float g = gptr[0];

  // O[i,c] = sum_j att[i,j] * V[c,j] ; V staged in 64-channel chunks [c][j]
#pragma unroll 1
  for (int cc = 0; cc < 4; ++cc) {
    // 2 j-positions x 8 channels per thread chunk, packed b32 LDS stores
#pragma unroll
    for (int chunk = 0; chunk < 2; ++chunk) {
      int c512 = tid + chunk * 256;                // 512 chunks of 16 elems
      int j0 = (c512 >> 3) * 2;                    // even j
      int cg = (c512 & 7) * 8;
      const __bf16* gsrc =
          vb + ((size_t)b * CC_HW + poff + (size_t)j0 * pstr) * CC_C +
          cc * 64 + cg;
      v8bf a8 = *(const v8bf*)gsrc;
      v8bf b8 = *(const v8bf*)(gsrc + (size_t)pstr * CC_C);
#pragma unroll
      for (int e = 0; e < 8; ++e)
        *(unsigned*)(stage + (cg + e) * 128 + j0) = pack2(a8[e], b8[e]);
    }
    __syncthreads();
#pragma unroll
    for (int ct = 0; ct < 4; ++ct) {
      v8f acc = {};
#pragma unroll
      for (int k2 = 0; k2 < 4; ++k2) {
        v16bf a2 = load_fragA(&att[(myi + lrow) * 128 + k2 * 32], koff);
        v16bf b2 = load_fragB(&stage[(ct * 16 + lrow) * 128 + k2 * 32], boff);
        acc = wmma_bf16(a2, b2, acc);
      }
      int c = cc * 64 + ct * 16 + lrow;
      if (MODE == 0) {                             // strided out_h stores
#pragma unroll
        for (int r = 0; r < 8; ++r) {
          int i = myi + r + Mhi;
          outp[(size_t)(b * CC_C + c) * CC_HW + (size_t)i * CC_W + fixed] =
              acc[r];
        }
      } else {                                     // contiguous v8f RMW
        size_t o = (size_t)(b * CC_C + c) * CC_HW + (size_t)fixed * CC_W +
                   myi + Mhi;
        v8f prev = *(v8f*)(outp + o);
        v8f xv   = *(const v8f*)(x + o);
        v8f res;
#pragma unroll
        for (int r = 0; r < 8; ++r) res[r] = g * (prev[r] + acc[r]) + xv[r];
        *(v8f*)(outp + o) = res;
      }
    }
    __syncthreads();
  }
}

// ---- launcher --------------------------------------------------------------

extern "C" void kernel_launch(void* const* d_in, const int* in_sizes, int n_in,
                              void* d_out, int out_size, void* d_ws,
                              size_t ws_size, hipStream_t stream) {
  const float* x     = (const float*)d_in[0];
  const float* Wq    = (const float*)d_in[1];
  const float* bq    = (const float*)d_in[2];
  const float* Wk    = (const float*)d_in[3];
  const float* bk    = (const float*)d_in[4];
  const float* Wv    = (const float*)d_in[5];
  const float* bv    = (const float*)d_in[6];
  const float* gamma = (const float*)d_in[7];
  float* outp = (float*)d_out;

  // workspace carve-up (bf16 elements)
  const size_t XB = (size_t)CC_B * CC_C * CC_HW;   // 33,554,432
  const size_t QB = (size_t)CC_B * CC_CQ * CC_HW;  //  4,194,304
  __bf16* ws  = (__bf16*)d_ws;
  __bf16* xb  = ws;                                // [b][c][pix]
  __bf16* qbf = ws + XB;                           // [b][pix][32]
  __bf16* kbf = ws + XB + QB;                      // [b][pix][32]
  __bf16* vbf = ws + XB + 2 * QB;                  // [b][pix][256]
  __bf16* wqb = ws + 2 * XB + 2 * QB;
  __bf16* wkb = wqb + (size_t)CC_CQ * CC_C;
  __bf16* wvb = wkb + (size_t)CC_CQ * CC_C;

  // 0) fp32 -> bf16 conversions
  cc_cvt_f32_bf16<<<4096, 256, 0, stream>>>(x, xb, (int)(XB / 4));
  cc_cvt_f32_bf16<<<8, 256, 0, stream>>>(Wq, wqb, CC_CQ * CC_C / 4);
  cc_cvt_f32_bf16<<<8, 256, 0, stream>>>(Wk, wkb, CC_CQ * CC_C / 4);
  cc_cvt_f32_bf16<<<64, 256, 0, stream>>>(Wv, wvb, CC_C * CC_C / 4);

  // 1) projections (q, k: M=32 -> NT=2, one chunk; v: M=256 -> NT=4, 4 chunks)
  dim3 gqk(CC_B * (CC_HW / 128), 1);
  dim3 gv(CC_B * (CC_HW / 128), 4);
  cc_proj_gemm<2, CC_CQ><<<gqk, 256, 0, stream>>>(wqb, bq, xb, qbf);
  cc_proj_gemm<2, CC_CQ><<<gqk, 256, 0, stream>>>(wkb, bk, xb, kbf);
  cc_proj_gemm<4, CC_C><<<gv, 256, 0, stream>>>(wvb, bv, xb, vbf);

  // 2) column attention -> out_h into d_out
  cc_attn<0><<<CC_B * CC_W, 256, 0, stream>>>(qbf, kbf, vbf, x, gamma, outp);
  // 3) row attention + fused gamma*(out_h+out_w)+x
  cc_attn<1><<<CC_B * CC_H, 256, 0, stream>>>(qbf, kbf, vbf, x, gamma, outp);
}
